// GIN_Layer_60653528154553
// MI455X (gfx1250) — matrix operand order repaired and checked by reading, not compiled, not run on hardware
//
#include <hip/hip_runtime.h>
#include <hip/hip_bf16.h>

#define N_NODES 100000
#define N_EDGES 1600000
#define D_IN    64
#define D_HID   128
#define BN_EPS  1e-5f

typedef __attribute__((ext_vector_type(16))) _Float16 v16h;
typedef __attribute__((ext_vector_type(8)))  _Float16 v8h;
typedef __attribute__((ext_vector_type(8)))  float    v8f;

#define CAT16(lo, hi) __builtin_shufflevector((lo), (hi), 0,1,2,3,4,5,6,7,8,9,10,11,12,13,14,15)

struct alignas(8) h4 { _Float16 v[4]; };

// ---------------------------------------------------------------------------
// Zero a float region.
__global__ void k_zero(float* __restrict__ p, long long n) {
    long long i = (long long)blockIdx.x * blockDim.x + threadIdx.x;
    if (i < n) p[i] = 0.0f;
}

// ---------------------------------------------------------------------------
// One-time weight transpose + f32->f16: WT[col][k] = (f16)W[k][col].
// Makes every WMMA B-fragment two contiguous 16-byte loads per lane.
__global__ void k_wcvt(const float* __restrict__ W1, const float* __restrict__ W2,
                       _Float16* __restrict__ W1T, _Float16* __restrict__ W2T) {
    int i = blockIdx.x * blockDim.x + threadIdx.x;
    if (i < D_IN * D_HID) {                 // W1T: [128][64]
        int col = i >> 6, k = i & 63;
        W1T[i] = (_Float16)W1[k * D_HID + col];
    } else if (i < D_IN * D_HID + D_HID * D_HID) {  // W2T: [128][128]
        int j = i - D_IN * D_HID;
        int col = j >> 7, k = j & 127;
        W2T[j] = (_Float16)W2[k * D_HID + col];
    }
}

// ---------------------------------------------------------------------------
// Edge scatter: agg[dst] += x[src].  16 threads per edge, float4 per thread.
__global__ void k_scatter(const float4* __restrict__ x4,
                          const int* __restrict__ ei,   // [2, N_EDGES] int32
                          float* __restrict__ agg) {
    long long tid = (long long)blockIdx.x * blockDim.x + threadIdx.x;
    if (tid >= (long long)N_EDGES * 16) return;
    int e = (int)(tid >> 4);
    int q = (int)(tid & 15);
    int s = ei[e];
    int d = ei[N_EDGES + e];
    float4 v = x4[(long long)s * 16 + q];
    float* dp = agg + (long long)d * D_IN + q * 4;
    atomicAdd(dp + 0, v.x);
    atomicAdd(dp + 1, v.y);
    atomicAdd(dp + 2, v.z);
    atomicAdd(dp + 3, v.w);
}

// ---------------------------------------------------------------------------
// Fused: h = x+agg -> f16 -> WMMA GEMM1 + b1 + ReLU -> LDS ->
//        WMMA GEMM2 + b2 + ReLU -> out, plus per-channel sum / sum-of-squares.
// Block = 128 threads (4 wave32), 64 rows per block.
__global__ void __launch_bounds__(128)
k_gin_mlp(const float* __restrict__ x, const float* __restrict__ agg,
          const _Float16* __restrict__ W1T,   // [128][64]  (col-major of W1)
          const float* __restrict__ b1,
          const _Float16* __restrict__ W2T,   // [128][128] (col-major of W2)
          const float* __restrict__ b2,
          float* __restrict__ out,
          float* __restrict__ sums, float* __restrict__ sumsq) {
    __shared__ _Float16 sA[64][D_IN];      // 8 KB  : (x+agg) tile, f16
    __shared__ _Float16 sH1[64][D_HID];    // 16 KB : hidden activations, f16
    __shared__ float    sSum[D_HID];       // per-channel partial sums
    __shared__ float    sSq[D_HID];

    const int tid  = threadIdx.x;          // 0..127
    const int wave = tid >> 5;             // 0..3  -> row strip wave*16..+15
    const int lane = tid & 31;
    const int half = lane >> 4;            // 0/1
    const int l16  = lane & 15;
    const long long rowBase = (long long)blockIdx.x * 64;

    sSum[tid] = 0.0f;
    sSq[tid]  = 0.0f;

    // Load h = x + agg -> f16 LDS, float4-vectorized (zero-pad tail rows).
    for (int i = tid; i < 64 * (D_IN / 4); i += 128) {
        int r = i >> 4, c4 = (i & 15) * 4;
        long long gr = rowBase + r;
        h4 o;
        if (gr < N_NODES) {
            long long o4 = gr * (D_IN / 4) + (i & 15);
            float4 xv = ((const float4*)x)[o4];
            float4 av = ((const float4*)agg)[o4];
            o.v[0] = (_Float16)(xv.x + av.x);
            o.v[1] = (_Float16)(xv.y + av.y);
            o.v[2] = (_Float16)(xv.z + av.z);
            o.v[3] = (_Float16)(xv.w + av.w);
        } else {
            o.v[0] = o.v[1] = o.v[2] = o.v[3] = (_Float16)0.0f;
        }
        *(h4*)&sA[r][c4] = o;
    }
    __syncthreads();

    // ---- GEMM1: [16 x 64] * [64 x 128], K-chunks of 32 ----
    const int arow = wave * 16 + l16;      // A-frag: lane holds row (l&15)
    // A-frag element i -> K = kc*32 + (i>=8?16:0) + half*8 + (i&7):
    // two contiguous 8-element runs -> two ds_load_b128 each.
    v16h a0 = CAT16(*(const v8h*)&sA[arow][half * 8],
                    *(const v8h*)&sA[arow][16 + half * 8]);
    v16h a1 = CAT16(*(const v8h*)&sA[arow][32 + half * 8],
                    *(const v8h*)&sA[arow][48 + half * 8]);
#pragma unroll
    for (int nt = 0; nt < 8; ++nt) {
        const int col = nt * 16 + l16;
        // B-frag element i -> K = kc*32 + (i>=8?16:0) + half*8 + (i&7): matches A-frag.
        const _Float16* wrow = W1T + col * D_IN;
        v16h bb0 = CAT16(*(const v8h*)(wrow + half * 8),
                         *(const v8h*)(wrow + 16 + half * 8));
        v16h bb1 = CAT16(*(const v8h*)(wrow + 32 + half * 8),
                         *(const v8h*)(wrow + 48 + half * 8));
        v8f c = {};
        c = __builtin_amdgcn_wmma_f32_16x16x32_f16(false, a0, false, bb0,
                                                   (short)0, c, false, false);
        c = __builtin_amdgcn_wmma_f32_16x16x32_f16(false, a1, false, bb1,
                                                   (short)0, c, false, false);
        float bias = b1[col];
#pragma unroll
        for (int r = 0; r < 8; ++r) {      // D: M = r + 8*half, N = col
            float v = c[r] + bias;
            v = v > 0.0f ? v : 0.0f;       // ReLU
            sH1[wave * 16 + r + 8 * half][col] = (_Float16)v;
        }
    }
    __syncthreads();

    // ---- GEMM2: [16 x 128] * [128 x 128], 4 K-chunks of 32 ----
    v16h a2[4];
#pragma unroll
    for (int kc = 0; kc < 4; ++kc)
        a2[kc] = CAT16(*(const v8h*)&sH1[arow][kc * 32 + half * 8],
                       *(const v8h*)&sH1[arow][kc * 32 + 16 + half * 8]);
#pragma unroll
    for (int nt = 0; nt < 8; ++nt) {
        const int col = nt * 16 + l16;
        const _Float16* wrow = W2T + col * D_HID;
        v8f c = {};
#pragma unroll
        for (int kc = 0; kc < 4; ++kc) {
            v16h bb = CAT16(*(const v8h*)(wrow + kc * 32 + half * 8),
                            *(const v8h*)(wrow + kc * 32 + 16 + half * 8));
            c = __builtin_amdgcn_wmma_f32_16x16x32_f16(false, a2[kc], false, bb,
                                                       (short)0, c, false, false);
        }
        float bias = b2[col];
#pragma unroll
        for (int r = 0; r < 8; ++r) {
            long long gr = rowBase + wave * 16 + r + 8 * half;
            float v = c[r] + bias;
            v = v > 0.0f ? v : 0.0f;       // outer ReLU
            if (gr < N_NODES) {
                out[gr * D_HID + col] = v;
                atomicAdd(&sSum[col], v);
                atomicAdd(&sSq[col], v * v);
            }
        }
    }
    __syncthreads();

    // One atomic per channel per block into global stats.
    atomicAdd(&sums[tid],  sSum[tid]);
    atomicAdd(&sumsq[tid], sSq[tid]);
}

// ---------------------------------------------------------------------------
// Finalize batchnorm: per-channel scale / shift.
__global__ void k_stats(const float* __restrict__ sums,
                        const float* __restrict__ sumsq,
                        const float* __restrict__ gamma,
                        const float* __restrict__ beta,
                        float* __restrict__ ss) {   // ss[0..127]=scale, [128..255]=shift
    int c = threadIdx.x;
    const float invN = 1.0f / (float)N_NODES;
    float mean = sums[c] * invN;
    float var  = sumsq[c] * invN - mean * mean;
    var = var > 0.0f ? var : 0.0f;
    float sc = gamma[c] * rsqrtf(var + BN_EPS);
    ss[c]         = sc;
    ss[D_HID + c] = beta[c] - mean * sc;
}

// ---------------------------------------------------------------------------
// Apply batchnorm in place on out (float4-coalesced).
__global__ void k_norm(float4* __restrict__ out4, const float* __restrict__ ss) {
    long long i = (long long)blockIdx.x * blockDim.x + threadIdx.x;
    if (i >= (long long)N_NODES * (D_HID / 4)) return;
    int c4 = (int)(i & (D_HID / 4 - 1)) * 4;
    float4 v = out4[i];
    v.x = v.x * ss[c4 + 0] + ss[D_HID + c4 + 0];
    v.y = v.y * ss[c4 + 1] + ss[D_HID + c4 + 1];
    v.z = v.z * ss[c4 + 2] + ss[D_HID + c4 + 2];
    v.w = v.w * ss[c4 + 3] + ss[D_HID + c4 + 3];
    out4[i] = v;
}

// ---------------------------------------------------------------------------
extern "C" void kernel_launch(void* const* d_in, const int* in_sizes, int n_in,
                              void* d_out, int out_size, void* d_ws, size_t ws_size,
                              hipStream_t stream) {
    const float* x     = (const float*)d_in[0];
    const int*   ei    = (const int*)  d_in[1];   // [2, N_EDGES] (JAX default int32)
    const float* W1    = (const float*)d_in[2];
    const float* b1    = (const float*)d_in[3];
    const float* W2    = (const float*)d_in[4];
    const float* b2    = (const float*)d_in[5];
    const float* gamma = (const float*)d_in[6];
    const float* beta  = (const float*)d_in[7];
    float* out = (float*)d_out;

    // Workspace layout
    float* agg   = (float*)d_ws;                          // N_NODES*D_IN floats
    float* sums  = agg + (long long)N_NODES * D_IN;       // D_HID
    float* sumsq = sums + D_HID;                          // D_HID
    float* ss    = sumsq + D_HID;                         // 2*D_HID
    _Float16* W1T = (_Float16*)(ss + 2 * D_HID);          // 128*64 f16 (16B-aligned)
    _Float16* W2T = W1T + D_IN * D_HID;                   // 128*128 f16

    // 1) zero agg + stats (contiguous region)
    {
        long long n = (long long)N_NODES * D_IN + 2 * D_HID;
        int blocks = (int)((n + 255) / 256);
        k_zero<<<blocks, 256, 0, stream>>>(agg, n);
    }
    // 1b) one-time weight transpose/convert to f16 fragment-friendly layout
    {
        int n = D_IN * D_HID + D_HID * D_HID;   // 24576
        k_wcvt<<<(n + 255) / 256, 256, 0, stream>>>(W1, W2, W1T, W2T);
    }
    // 2) edge scatter-sum (float atomics; agg fits in L2)
    {
        long long n = (long long)N_EDGES * 16;
        int blocks = (int)((n + 255) / 256);
        k_scatter<<<blocks, 256, 0, stream>>>((const float4*)x, ei, agg);
    }
    // 3) fused (x+agg) -> MLP via WMMA -> out + stats partials
    {
        int blocks = (N_NODES + 63) / 64;   // 1563
        k_gin_mlp<<<blocks, 128, 0, stream>>>(x, agg, W1T, b1, W2T, b2,
                                              out, sums, sumsq);
    }
    // 4) finalize batchnorm scale/shift
    k_stats<<<1, D_HID, 0, stream>>>(sums, sumsq, gamma, beta, ss);
    // 5) normalize in place
    {
        long long n = (long long)N_NODES * (D_HID / 4);
        int blocks = (int)((n + 255) / 256);
        k_norm<<<blocks, 256, 0, stream>>>((float4*)out, ss);
    }
}
